// MultiHeadAttention_16114717295169
// MI455X (gfx1250) — matrix operand order repaired and checked by reading, not compiled, not run on hardware
//
#include <hip/hip_runtime.h>
#include <hip/hip_bf16.h>
#include <math.h>

// ---------------------------------------------------------------------------
// Graphormer-style MHA for MI455X (gfx1250), wave32 + WMMA f32_16x16x32_f16.
// S=1024, B=8, E=512, H=8, D=64, N=B*H=64.
//
// Pipeline (all on `stream`, ws needs 320 MiB):
//  K1 qkv   : q/k/v = x@W^T+b  (WMMA), f16 outputs q_h,k_h,v_h [n][s][d] and
//             q_perm [s][n][d]
//  K2 ebias : per-j GEMM Q_j(64x64) @ EK_j^T(64x1024) -> L[n][i][j] f16,
//             wave-private LDS staging for coalesced tiled writes
//  K3 attn  : per (n, 16 rows): S=QK^T (WMMA) into LDS rows, add bias row
//             (coalesced L read), mask, softmax (shfl half-row reduce),
//             normalized P -> LDS f16 + coalesced global P2 write,
//             out1 = P@V (WMMA, A-frags from LDS)
//  K4 eout  : per-i GEMM P[:,i,:](64x1024) @ EV_i(1024x64) via LDS-staged
//             tiles; P tiles staged with ASYNC global->LDS copies
//             (global_load_async_to_lds_b128, ASYNCcnt tracked)
//  K5 oproj : x = context@Wo^T + bo + residual (WMMA)
//  K6 ln    : LayerNorm over E=512
// ---------------------------------------------------------------------------

typedef _Float16 f16;
typedef __attribute__((ext_vector_type(16))) _Float16 v16h;
typedef __attribute__((ext_vector_type(8)))  _Float16 v8h;
typedef __attribute__((ext_vector_type(8)))  float    v8f;
typedef __attribute__((ext_vector_type(4)))  float    v4f;
typedef int v4i __attribute__((vector_size(16)));

#if __has_builtin(__builtin_amdgcn_global_load_async_to_lds_b128) && \
    __has_builtin(__builtin_amdgcn_s_wait_asynccnt)
#define USE_ASYNC_LDS 1
#else
#define USE_ASYNC_LDS 0
#endif

#if USE_ASYNC_LDS
typedef __attribute__((address_space(1))) v4i* gv4i_p;  // global int4*
typedef __attribute__((address_space(3))) v4i* lv4i_p;  // LDS int4*
static __device__ __forceinline__ void async_copy16(const void* g, void* l) {
  __builtin_amdgcn_global_load_async_to_lds_b128((gv4i_p)g, (lv4i_p)l, 0, 0);
}
#endif

static __device__ __forceinline__ v8f wmma16(v16h a, v16h b, v8f c) {
  return __builtin_amdgcn_wmma_f32_16x16x32_f16(false, a, false, b, (short)0, c,
                                                false, false);
}

// A-matrix 16x32 f16 fragment: lane holds row (l&15); K = e + (l&16)
static __device__ __forceinline__ v16h fragA_h(const f16* rp, int k0, int l) {
  const int off = l & 16;
  v8h lo = *(const v8h*)(rp + k0 + off);
  v8h hi = *(const v8h*)(rp + k0 + off + 8);
  v16h r;
#pragma unroll
  for (int q = 0; q < 8; ++q) { r[q] = lo[q]; r[8 + q] = hi[q]; }
  return r;
}
static __device__ __forceinline__ v16h fragA_f32(const float* rp, int k0, int l) {
  const int off = l & 16;
  v4f a0 = *(const v4f*)(rp + k0 + off);
  v4f a1 = *(const v4f*)(rp + k0 + off + 4);
  v4f a2 = *(const v4f*)(rp + k0 + off + 8);
  v4f a3 = *(const v4f*)(rp + k0 + off + 12);
  v16h r;
#pragma unroll
  for (int q = 0; q < 4; ++q) {
    r[q] = (f16)a0[q]; r[4 + q] = (f16)a1[q];
    r[8 + q] = (f16)a2[q]; r[12 + q] = (f16)a3[q];
  }
  return r;
}
// B-matrix 32x16 f16 fragment: lane holds col (l&15); K = e + (l&16)
static __device__ __forceinline__ v16h fragB_h(const f16* rp, int k0, int l) {
  const int off = l & 16;
  v8h lo = *(const v8h*)(rp + k0 + off);
  v8h hi = *(const v8h*)(rp + k0 + off + 8);
  v16h r;
#pragma unroll
  for (int q = 0; q < 8; ++q) { r[q] = lo[q]; r[8 + q] = hi[q]; }
  return r;
}
static __device__ __forceinline__ v16h fragB_f32(const float* rp, int k0, int l) {
  const int off = l & 16;
  v4f a0 = *(const v4f*)(rp + k0 + off);
  v4f a1 = *(const v4f*)(rp + k0 + off + 4);
  v4f a2 = *(const v4f*)(rp + k0 + off + 8);
  v4f a3 = *(const v4f*)(rp + k0 + off + 12);
  v16h r;
#pragma unroll
  for (int q = 0; q < 4; ++q) {
    r[q] = (f16)a0[q]; r[4 + q] = (f16)a1[q];
    r[8 + q] = (f16)a2[q]; r[12 + q] = (f16)a3[q];
  }
  return r;
}

// ============================ K1: QKV projections ==========================
// grid (512, 4, 3), block 256. wave w handles output cols n0=(by*8+w)*16.
__global__ __launch_bounds__(256) void k_qkv(
    const float* __restrict__ q, const float* __restrict__ k,
    const float* __restrict__ v,
    const float* __restrict__ Wq, const float* __restrict__ bq,
    const float* __restrict__ Wk, const float* __restrict__ bk,
    const float* __restrict__ Wv, const float* __restrict__ bv,
    f16* __restrict__ qh, f16* __restrict__ kh, f16* __restrict__ vh,
    f16* __restrict__ qp) {
  const int tid = threadIdx.x, w = tid >> 5, l = tid & 31;
  const int z = blockIdx.z;
  const float* X = (z == 0) ? q : (z == 1) ? k : v;
  const float* W = (z == 0) ? Wq : (z == 1) ? Wk : Wv;
  const float* BI = (z == 0) ? bq : (z == 1) ? bk : bv;
  f16* dst = (z == 0) ? qh : (z == 1) ? kh : vh;

  const int m0 = blockIdx.x * 16;
  const int n0 = (blockIdx.y * 8 + w) * 16;
  const float* arow = X + (size_t)(m0 + (l & 15)) * 512;
  const float* brow = W + (size_t)(n0 + (l & 15)) * 512;

  v8f c = {};
#pragma unroll
  for (int k0 = 0; k0 < 512; k0 += 32)
    c = wmma16(fragA_f32(arow, k0, l), fragB_f32(brow, k0, l), c);

  const float bb = BI[n0 + (l & 15)];
  const int head = n0 >> 6;
  const int d = (n0 & 63) + (l & 15);
  const int hi = (l >> 4) << 3;
#pragma unroll
  for (int r = 0; r < 8; ++r) {
    const int m = m0 + r + hi;
    const int s = m >> 3, b_ = m & 7;
    const int n = b_ * 8 + head;
    const f16 hv = (f16)(c[r] + bb);
    dst[((size_t)n * 1024 + s) * 64 + d] = hv;
    if (z == 0) qp[((size_t)s * 64 + n) * 64 + d] = hv;
  }
}

// ============================ K2: edge-key bias ============================
// bias[n,i,j] = sum_d ek[j,i,d]*q[n,j,d]. Per j: Q_j(64x64) @ EK_j^T(64x1024).
// grid (64 jb, 8 ib), block 256, dyn LDS 256 KB (32 KB/wave staging).
__global__ __launch_bounds__(256) void k_edge_bias(
    const float* __restrict__ ek, const f16* __restrict__ qp,
    f16* __restrict__ Lb) {
  extern __shared__ char dynsm[];
  const int tid = threadIdx.x, w = tid >> 5, l = tid & 31;
  f16* buf = (f16*)dynsm + (size_t)w * 16384;  // [j 16][n 64][i 16]
  const int j0 = blockIdx.x * 16;
  const int i0w = blockIdx.y * 128 + w * 16;
  const int hi = (l >> 4) << 3;

  for (int jl = 0; jl < 16; ++jl) {
    const int j = j0 + jl;
    const float* ekrow = ek + ((size_t)j * 1024 + i0w + (l & 15)) * 64;
    __builtin_prefetch(ekrow + 1024 * 64, 0, 1);  // next j row, same i
    const v16h b0 = fragB_f32(ekrow, 0, l);
    const v16h b1 = fragB_f32(ekrow, 32, l);
#pragma unroll
    for (int nt = 0; nt < 4; ++nt) {
      const f16* qrow = qp + ((size_t)j * 64 + nt * 16 + (l & 15)) * 64;
      v8f c = {};
      c = wmma16(fragA_h(qrow, 0, l), b0, c);
      c = wmma16(fragA_h(qrow, 32, l), b1, c);
#pragma unroll
      for (int r = 0; r < 8; ++r)
        buf[((size_t)jl * 64 + nt * 16 + r + hi) * 16 + (l & 15)] = (f16)c[r];
    }
  }
  // wave-private transpose write-out: 16 j's per (n,i) -> 32B chunks
  for (int p = 0; p < 32; ++p) {
    const int id = p * 32 + l;
    const int n = id >> 4, il = id & 15;
    v8h h0, h1;
#pragma unroll
    for (int jl = 0; jl < 8; ++jl) {
      h0[jl] = buf[((size_t)jl * 64 + n) * 16 + il];
      h1[jl] = buf[((size_t)(jl + 8) * 64 + n) * 16 + il];
    }
    f16* Lp = Lb + ((size_t)n * 1024 + i0w + il) * 1024 + j0;
    *(v8h*)Lp = h0;
    *(v8h*)(Lp + 8) = h1;
  }
}

// ============================ K3: attention ================================
// grid (16, 64), block 128 (4 waves), dyn LDS 4*16*1032*4 = 264192 B.
__global__ __launch_bounds__(128) void k_attn(
    const f16* __restrict__ qh, const f16* __restrict__ kh,
    const f16* __restrict__ vh, const f16* __restrict__ Lb,
    const unsigned char* __restrict__ mask, f16* __restrict__ P2,
    float* __restrict__ oattn) {
  extern __shared__ char dynsm[];
  float* smf = (float*)dynsm;
  const int tid = threadIdx.x, w = tid >> 5, l = tid & 31;
  const int n = blockIdx.y;
  const int i0 = blockIdx.x * 64 + w * 16;
  float* S = smf + (size_t)w * 16 * 1032;  // 16 rows x 1032 f32 (wave-private)
  const int hi = (l >> 4) << 3;

  // ---- phase 1: S = Q K^T
  const f16* qrow = qh + ((size_t)n * 1024 + i0 + (l & 15)) * 64;
  const v16h a0 = fragA_h(qrow, 0, l), a1 = fragA_h(qrow, 32, l);
  for (int jt = 0; jt < 64; ++jt) {
    const f16* krow = kh + ((size_t)n * 1024 + jt * 16 + (l & 15)) * 64;
    v8f c = {};
    c = wmma16(a0, fragB_h(krow, 0, l), c);
    c = wmma16(a1, fragB_h(krow, 32, l), c);
#pragma unroll
    for (int r = 0; r < 8; ++r) S[(r + hi) * 1032 + jt * 16 + (l & 15)] = c[r];
  }

  // ---- phase 2: + bias, scale, mask, rowwise max & exp-sum (half-row/lane)
  const int r_ = l & 15, h_ = l >> 4;
  float* row = S + r_ * 1032;
  const f16* brow = Lb + ((size_t)n * 1024 + i0 + r_) * 1024;
  const unsigned char* mrow =
      mask + (size_t)n * 1048576 + (size_t)(i0 + r_) * 1024;
  const float scale = 0.125f;
  const float NEGB = -3.0e38f;
  float mx = NEGB;
  for (int j = h_ * 512; j < h_ * 512 + 512; j += 8) {
    const v8h bv = *(const v8h*)(brow + j);
    const unsigned long long mb = *(const unsigned long long*)(mrow + j);
    v4f x0 = *(v4f*)(row + j), x1 = *(v4f*)(row + j + 4);
#pragma unroll
    for (int qq = 0; qq < 8; ++qq) {
      float vv = ((qq < 4 ? x0[qq] : x1[qq - 4]) + (float)bv[qq]) * scale;
      if ((mb >> (8 * qq)) & 1ull) vv = NEGB;
      row[j + qq] = vv;
      mx = fmaxf(mx, vv);
    }
  }
  mx = fmaxf(mx, __shfl_xor(mx, 16, 32));
  float sum = 0.f;
  for (int j = h_ * 512; j < h_ * 512 + 512; ++j) {
    const float p = __expf(row[j] - mx);
    row[j] = p;
    sum += p;
  }
  sum += __shfl_xor(sum, 16, 32);
  const float inv = 1.0f / sum;

  // ---- phase 3: normalize -> f16 in-place (interleaved 8-chunks, race-free)
  f16* hrow = (f16*)((char*)smf + (size_t)(w * 16 + r_) * 4128);
  for (int kc = 0; kc < 64; ++kc) {
    const int jc = kc * 16 + h_ * 8;
    v4f x0 = *(v4f*)(row + jc), x1 = *(v4f*)(row + jc + 4);
    v8h ph;
#pragma unroll
    for (int qq = 0; qq < 8; ++qq)
      ph[qq] = (f16)((qq < 4 ? x0[qq] : x1[qq - 4]) * inv);
    *(v8h*)(hrow + jc) = ph;
  }

  // ---- phase 4: coalesced global write of P2[n][i][j]
  for (int rr = 0; rr < 16; ++rr) {
    const f16* src = (const f16*)((const char*)smf + (size_t)(w * 16 + rr) * 4128);
    f16* dst = P2 + ((size_t)n * 1024 + i0 + rr) * 1024 + l * 32;
#pragma unroll
    for (int qq = 0; qq < 4; ++qq)
      *(v8h*)(dst + qq * 8) = *(const v8h*)(src + l * 32 + qq * 8);
  }

  // ---- phase 5: out1 = P @ V
  const f16* prow = (const f16*)((const char*)smf + (size_t)(w * 16 + (l & 15)) * 4128);
  v8f o0 = {}, o1 = {}, o2 = {}, o3 = {};
  for (int ks = 0; ks < 32; ++ks) {
    const v16h a = fragA_h(prow, ks * 32, l);
    const int offb = l & 16;
#pragma unroll
    for (int dt = 0; dt < 4; ++dt) {
      v16h b;
#pragma unroll
      for (int e = 0; e < 16; ++e) {
        const int j = ks * 32 + e + offb;
        b[e] = vh[(size_t)n * 65536 + (size_t)j * 64 + dt * 16 + (l & 15)];
      }
      if (dt == 0) o0 = wmma16(a, b, o0);
      else if (dt == 1) o1 = wmma16(a, b, o1);
      else if (dt == 2) o2 = wmma16(a, b, o2);
      else o3 = wmma16(a, b, o3);
    }
  }
#pragma unroll
  for (int r = 0; r < 8; ++r) {
    const size_t base = ((size_t)n * 1024 + i0 + r + hi) * 64 + (l & 15);
    oattn[base] = o0[r];
    oattn[base + 16] = o1[r];
    oattn[base + 32] = o2[r];
    oattn[base + 48] = o3[r];
  }
}

// ============================ K4: edge-value ===============================
// per i: oattn[:,i,:] += P[:,i,:](64x1024) @ EV_i(1024x64). grid 1024, blk 256.
__global__ __launch_bounds__(256) void k_edge_out(
    const f16* __restrict__ P2, const float* __restrict__ ev,
    float* __restrict__ oattn) {
  __shared__ f16 Pl[64 * 72];   // [n][j_local], padded rows
  __shared__ f16 Evl[64 * 72];  // [j_local][d], padded rows
  const int tid = threadIdx.x, w = tid >> 5, l = tid & 31;
  const int i = blockIdx.x;
  const int nt = w >> 1, dtb = (w & 1) * 2;
  v8f c0 = {}, c1 = {};

  for (int jc = 0; jc < 16; ++jc) {
    const int j0 = jc * 64;
    __syncthreads();
    {  // cooperative stage: P rows (async f16 copy) and EV rows (f32->f16)
      const int rowi = tid >> 2, q = tid & 3;
      const f16* ps = P2 + ((size_t)rowi * 1024 + i) * 1024 + j0 + q * 16;
      f16* pd = Pl + rowi * 72 + q * 16;
#if USE_ASYNC_LDS
      async_copy16(ps, pd);           // global_load_async_to_lds_b128
      async_copy16(ps + 8, pd + 8);   // (ASYNCcnt-tracked, no VGPR staging)
#else
      *(v8h*)pd = *(const v8h*)ps;
      *(v8h*)(pd + 8) = *(const v8h*)(ps + 8);
#endif
      const float* es = ev + ((size_t)i * 1024 + j0 + rowi) * 64 + q * 16;
      if (jc + 1 < 16) __builtin_prefetch(es + 64 * 64, 0, 1);
      v4f e0 = *(const v4f*)es, e1 = *(const v4f*)(es + 4);
      v4f e2 = *(const v4f*)(es + 8), e3 = *(const v4f*)(es + 12);
      v8h ha, hb;
#pragma unroll
      for (int qq = 0; qq < 4; ++qq) {
        ha[qq] = (f16)e0[qq]; ha[4 + qq] = (f16)e1[qq];
        hb[qq] = (f16)e2[qq]; hb[4 + qq] = (f16)e3[qq];
      }
      *(v8h*)(Evl + rowi * 72 + q * 16) = ha;
      *(v8h*)(Evl + rowi * 72 + q * 16 + 8) = hb;
#if USE_ASYNC_LDS
      __builtin_amdgcn_s_wait_asynccnt(0);  // this wave's async copies done
#endif
    }
    __syncthreads();
#pragma unroll
    for (int ks = 0; ks < 2; ++ks) {
      const v16h a = fragA_h(Pl + (nt * 16 + (l & 15)) * 72, ks * 32, l);
      const int offb = l & 16;
#pragma unroll
      for (int dd = 0; dd < 2; ++dd) {
        const int d = (dtb + dd) * 16 + (l & 15);
        v16h b;
#pragma unroll
        for (int e = 0; e < 16; ++e)
          b[e] = Evl[(ks * 32 + e + offb) * 72 + d];
        if (dd == 0) c0 = wmma16(a, b, c0);
        else c1 = wmma16(a, b, c1);
      }
    }
  }
  const int hi = (l >> 4) << 3;
#pragma unroll
  for (int r = 0; r < 8; ++r) {
    const size_t base =
        ((size_t)(nt * 16 + r + hi) * 1024 + i) * 64 + dtb * 16 + (l & 15);
    oattn[base] += c0[r];
    oattn[base + 16] += c1[r];
  }
}

// ============================ K5: output projection ========================
// grid (512, 4), block 256. x = context @ Wo^T + bo + residual.
__global__ __launch_bounds__(256) void k_oproj(
    const float* __restrict__ oattn, const float* __restrict__ Wo,
    const float* __restrict__ bo, const float* __restrict__ resid,
    float* __restrict__ xbuf) {
  const int tid = threadIdx.x, w = tid >> 5, l = tid & 31;
  const int m0 = blockIdx.x * 16;
  const int n0 = (blockIdx.y * 8 + w) * 16;
  const int m = m0 + (l & 15);
  const int s = m >> 3, b_ = m & 7;
  const float* brow = Wo + (size_t)(n0 + (l & 15)) * 512;

  v8f c = {};
#pragma unroll
  for (int k0 = 0; k0 < 512; k0 += 32) {
    const int off = l & 16;
    v16h a;
    {
      const int kk = k0 + off, hd = kk >> 6, dd = kk & 63;
      const float* p = oattn + ((size_t)(b_ * 8 + hd) * 1024 + s) * 64 + dd;
      v4f x0 = *(const v4f*)p, x1 = *(const v4f*)(p + 4);
#pragma unroll
      for (int qq = 0; qq < 4; ++qq) { a[qq] = (f16)x0[qq]; a[4 + qq] = (f16)x1[qq]; }
    }
    {
      const int kk = k0 + off + 8, hd = kk >> 6, dd = kk & 63;
      const float* p = oattn + ((size_t)(b_ * 8 + hd) * 1024 + s) * 64 + dd;
      v4f x0 = *(const v4f*)p, x1 = *(const v4f*)(p + 4);
#pragma unroll
      for (int qq = 0; qq < 4; ++qq) { a[8 + qq] = (f16)x0[qq]; a[12 + qq] = (f16)x1[qq]; }
    }
    c = wmma16(a, fragB_f32(brow, k0, l), c);
  }
  const float bb = bo[n0 + (l & 15)];
  const int hi = (l >> 4) << 3;
#pragma unroll
  for (int r = 0; r < 8; ++r) {
    const int mm = m0 + r + hi;
    const size_t idx = (size_t)mm * 512 + n0 + (l & 15);
    xbuf[idx] = c[r] + bb + resid[idx];
  }
}

// ============================ K6: LayerNorm ================================
__global__ __launch_bounds__(256) void k_ln(
    const float* __restrict__ x, const float* __restrict__ gamma,
    const float* __restrict__ beta, float* __restrict__ out) {
  __shared__ float ws1[8], ws2[8];
  const int row = blockIdx.x, t = threadIdx.x;
  const float a = x[(size_t)row * 512 + t];
  const float b = x[(size_t)row * 512 + 256 + t];
  float s = a + b, s2 = a * a + b * b;
#pragma unroll
  for (int o = 16; o >= 1; o >>= 1) {
    s += __shfl_xor(s, o, 32);
    s2 += __shfl_xor(s2, o, 32);
  }
  if ((t & 31) == 0) { ws1[t >> 5] = s; ws2[t >> 5] = s2; }
  __syncthreads();
  float ts = 0.f, ts2 = 0.f;
#pragma unroll
  for (int q = 0; q < 8; ++q) { ts += ws1[q]; ts2 += ws2[q]; }
  const float mean = ts * (1.0f / 512.0f);
  const float var = ts2 * (1.0f / 512.0f) - mean * mean;
  const float inv = rsqrtf(var + 1e-5f);
  out[(size_t)row * 512 + t] = gamma[t] * ((a - mean) * inv) + beta[t];
  out[(size_t)row * 512 + 256 + t] =
      gamma[256 + t] * ((b - mean) * inv) + beta[256 + t];
}

// ============================ host launcher ================================
extern "C" void kernel_launch(void* const* d_in, const int* in_sizes, int n_in,
                              void* d_out, int out_size, void* d_ws,
                              size_t ws_size, hipStream_t stream) {
  const float* query = (const float*)d_in[0];
  const float* key = (const float*)d_in[1];
  const float* value = (const float*)d_in[2];
  const float* ek = (const float*)d_in[3];
  const float* ev = (const float*)d_in[4];
  const unsigned char* mask = (const unsigned char*)d_in[5];
  const float* Wq = (const float*)d_in[7];
  const float* bq = (const float*)d_in[8];
  const float* Wk = (const float*)d_in[9];
  const float* bk = (const float*)d_in[10];
  const float* Wv = (const float*)d_in[11];
  const float* bv = (const float*)d_in[12];
  const float* Wo = (const float*)d_in[13];
  const float* bo = (const float*)d_in[14];
  const float* gamma = (const float*)d_in[15];
  const float* beta = (const float*)d_in[16];
  float* out = (float*)d_out;

  // workspace layout (bytes), total 320 MiB
  char* ws = (char*)d_ws;
  f16* qh = (f16*)(ws + (0ull << 20));
  f16* kh = (f16*)(ws + (8ull << 20));
  f16* vh = (f16*)(ws + (16ull << 20));
  f16* qp = (f16*)(ws + (24ull << 20));
  float* oattn = (float*)(ws + (32ull << 20));
  float* xbuf = (float*)(ws + (48ull << 20));
  f16* Lb = (f16*)(ws + (64ull << 20));
  f16* P2 = (f16*)(ws + (192ull << 20));

  // opt in to >64KB dynamic LDS (CDNA5 WGP has 320KB)
  (void)hipFuncSetAttribute(reinterpret_cast<const void*>(&k_edge_bias),
                            hipFuncAttributeMaxDynamicSharedMemorySize, 262144);
  (void)hipFuncSetAttribute(reinterpret_cast<const void*>(&k_attn),
                            hipFuncAttributeMaxDynamicSharedMemorySize, 264192);

  k_qkv<<<dim3(512, 4, 3), 256, 0, stream>>>(query, key, value, Wq, bq, Wk, bk,
                                             Wv, bv, qh, kh, vh, qp);
  k_edge_bias<<<dim3(64, 8), 256, 262144, stream>>>(ek, qp, Lb);
  k_attn<<<dim3(16, 64), 128, 264192, stream>>>(qh, kh, vh, Lb, mask, P2,
                                                oattn);
  k_edge_out<<<dim3(1024), 256, 0, stream>>>(P2, ev, oattn);
  k_oproj<<<dim3(512, 4), 256, 0, stream>>>(oattn, Wo, bo, query, xbuf);
  k_ln<<<dim3(8192), 256, 0, stream>>>(xbuf, gamma, beta, out);
}
